// GNN_MLP_60902636257640
// MI455X (gfx1250) — matrix operand order repaired and checked by reading, not compiled, not run on hardware
//
#include <hip/hip_runtime.h>

#define NODES_ 14
#define IN_F_ 24
#define HID1_ 32
#define HID2_ 64
#define HEADS_ 2
#define OUT_F_ 24
#define NB_ 2048
#define NTOT_ (NB_ * NODES_)   // 28672 nodes

typedef __attribute__((ext_vector_type(16))) _Float16 v16h;
typedef __attribute__((ext_vector_type(8)))  float    v8f;

// ---- WMMA fragment loaders (CDNA5 16x16x32 f16 layouts, wave32) ----------
// Branch-free: padded positions use a clamped (always in-bounds) address and
// a post-load select -> v_cndmask, never exec-masked scalar loads.

// A: 16xK tile, row-major f32 source, converted to f16.
// lane<16: M=lane, K in [0..7] and [16..23]; lane>=16: K in [8..15] and [24..31].
template <int KMAX>
__device__ __forceinline__ v16h load_a_tile(const float* __restrict__ base, int lda) {
  const int lane = threadIdx.x & 31;
  const int m  = lane & 15;
  const int kb = (lane >> 4) << 3;   // 0 or 8
  v16h a;
#pragma unroll
  for (int i = 0; i < 8; ++i) {
    const int k0 = kb + i;
    const int k1 = kb + 16 + i;
    const int c0 = (k0 < KMAX) ? k0 : (KMAX - 1);
    const int c1 = (k1 < KMAX) ? k1 : (KMAX - 1);
    const float v0 = base[m * lda + c0];
    const float v1 = base[m * lda + c1];
    a[i]     = (k0 < KMAX) ? (_Float16)v0 : (_Float16)0.0f;
    a[i + 8] = (k1 < KMAX) ? (_Float16)v1 : (_Float16)0.0f;
  }
  return a;
}

// B: 32x16 (KxN) tile from row-major [K x ldb] f32 weights.
// lane<16: N=lane, K in [0..7] and [16..23]; lane>=16: N=lane-16, K in [8..15] and [24..31].
template <int KMAX, int NVALID>
__device__ __forceinline__ v16h load_b_tile(const float* __restrict__ w, int ldb,
                                            int ncol0) {
  const int lane = threadIdx.x & 31;
  const int n  = lane & 15;
  const int nc = (n < NVALID) ? n : (NVALID - 1);
  const int kb = (lane >> 4) << 3;   // 0 or 8
  v16h b;
#pragma unroll
  for (int i = 0; i < 16; ++i) {
    const int k = (i < 8) ? (kb + i) : (kb + 8 + i);
    const int kc = (k < KMAX) ? k : (KMAX - 1);
    const float v = w[kc * ldb + (ncol0 + nc)];
    b[i] = (k < KMAX && n < NVALID) ? (_Float16)v : (_Float16)0.0f;
  }
  return b;
}

// ---- Projection: XH[N x NOUT] = X[N x KIN] @ W[KIN x NOUT] ---------------
template <int KIN, int NOUT>
__global__ void __launch_bounds__(256)
k_proj(const float* __restrict__ X, const float* __restrict__ W,
       float* __restrict__ XH) {
  constexpr int TILES = NTOT_ / 16;
  const int wave = threadIdx.x >> 5;
  const int tile = blockIdx.x * 8 + wave;
  if (tile >= TILES) return;   // grid sized exactly; never taken
  const int lane = threadIdx.x & 31;
  const v16h a = load_a_tile<KIN>(X + (size_t)tile * 16 * KIN, KIN);
  const int n  = lane & 15;
  const int mo = (lane >> 4) << 3;   // C rows: v -> v + mo
  float* obase = XH + (size_t)tile * 16 * NOUT;
#pragma unroll
  for (int nt = 0; nt < NOUT / 16; ++nt) {
    const v16h b = load_b_tile<KIN, 16>(W, NOUT, nt * 16);
    v8f c = {};
    c = __builtin_amdgcn_wmma_f32_16x16x32_f16(false, a, false, b,
                                               (short)0, c, false, false);
#pragma unroll
    for (int v = 0; v < 8; ++v)
      obase[(size_t)(v + mo) * NOUT + nt * 16 + n] = c[v];
  }
}

// ---- GAT attention: dense 14-wide softmax + aggregate per graph ----------
// One thread = one destination node. GA graphs per block (GA*14 threads).
template <int F, int GA>
__global__ void k_attn(const float* __restrict__ XH, const float* __restrict__ ASRC,
                       const float* __restrict__ ADST, const float* __restrict__ BIAS,
                       float* __restrict__ HOUT, int apply_elu) {
  constexpr int NT = GA * NODES_;
  __shared__ float s_xh[NT * HEADS_ * F];
  __shared__ float s_als[NT][HEADS_];
  __shared__ float s_ald[NT][HEADS_];
  const int t = threadIdx.x;
  const size_t base = (size_t)blockIdx.x * NT * HEADS_ * F;
  constexpr int TOT4 = NT * HEADS_ * F / 4;
  const float4* s4 = (const float4*)(XH + base);
  float4* d4 = (float4*)s_xh;
  for (int i = t; i < TOT4; i += NT) d4[i] = s4[i];
  __syncthreads();
  // attention logits per (node, head)
#pragma unroll
  for (int h = 0; h < HEADS_; ++h) {
    float ss = 0.f, sd = 0.f;
    for (int f = 0; f < F; ++f) {
      const float v = s_xh[(t * HEADS_ + h) * F + f];
      ss += v * ASRC[h * F + f];
      sd += v * ADST[h * F + f];
    }
    s_als[t][h] = ss;
    s_ald[t][h] = sd;
  }
  __syncthreads();
  const int gbase = (t / NODES_) * NODES_;
  float acc[F];
#pragma unroll
  for (int f = 0; f < F; ++f) acc[f] = 0.f;
  for (int h = 0; h < HEADS_; ++h) {
    const float ald = s_ald[t][h];
    float al[NODES_];
    float amax = -3.4e38f;
#pragma unroll
    for (int s = 0; s < NODES_; ++s) {
      float a = s_als[gbase + s][h] + ald;
      a = (a > 0.f) ? a : 0.2f * a;            // leaky_relu(0.2)
      al[s] = a;
      amax = fmaxf(amax, a);
    }
    float den = 0.f;
#pragma unroll
    for (int s = 0; s < NODES_; ++s) { al[s] = __expf(al[s] - amax); den += al[s]; }
    const float inv = 1.f / den;
    for (int s = 0; s < NODES_; ++s) {
      const float w = al[s] * inv;
      const float* xs = &s_xh[((gbase + s) * HEADS_ + h) * F];
#pragma unroll
      for (int f = 0; f < F; ++f) acc[f] += w * xs[f];
    }
  }
  float* orow = HOUT + ((size_t)blockIdx.x * NT + t) * F;
  for (int f = 0; f < F; ++f) {
    float v = acc[f] * 0.5f + BIAS[f];          // mean over 2 heads + bias
    if (apply_elu) v = (v > 0.f) ? v : (__expf(v) - 1.f);
    orow[f] = v;
  }
}

// ---- Per-node-index MLP: sigmoid(relu(h2_k @ Wf1[k] + bf1) @ Wf2[k] + bf2)
// One wave = (node index k, tile of 16 graphs). 14*128 = 1792 waves.
__global__ void __launch_bounds__(256)
k_mlp(const float* __restrict__ H2, const float* __restrict__ WF1,
      const float* __restrict__ BF1, const float* __restrict__ WF2,
      const float* __restrict__ BF2, float* __restrict__ OUT) {
  __shared__ float s_hh[8][16][8];
  const int wave = threadIdx.x >> 5;
  const int lane = threadIdx.x & 31;
  const int idx = blockIdx.x * 8 + wave;   // 0..1791
  const int k  = idx >> 7;                 // node index 0..13
  const int gt = idx & 127;                // graph tile 0..127
  const int m   = lane & 15;
  const int kb8 = (lane >> 4) << 3;
  const float* wf1k = WF1 + k * HID2_ * 8;
  v8f c = {};
#pragma unroll
  for (int kt = 0; kt < 2; ++kt) {         // K = 64 -> two 16x16x32 WMMAs
    const float* row = H2 + ((size_t)(gt * 16 + m) * NODES_ + k) * HID2_ + kt * 32;
    v16h a;
#pragma unroll
    for (int i = 0; i < 8; ++i) {          // all K valid: contiguous loads
      a[i]     = (_Float16)row[kb8 + i];
      a[i + 8] = (_Float16)row[kb8 + 16 + i];
    }
    const v16h b = load_b_tile<32, 8>(wf1k + kt * 32 * 8, 8, 0);
    c = __builtin_amdgcn_wmma_f32_16x16x32_f16(false, a, false, b,
                                               (short)0, c, false, false);
  }
  const int n  = lane & 15;
  const int mo = (lane >> 4) << 3;
  if (n < 8) {
#pragma unroll
    for (int v = 0; v < 8; ++v) {
      float hv = c[v] + BF1[k * 8 + n];
      s_hh[wave][v + mo][n] = (hv > 0.f) ? hv : 0.f;   // relu
    }
  }
  __syncthreads();
  // tiny second layer on VALU: 16 rows x 24 outputs, K=8
  const int m2 = lane & 15;
  const int ph = (lane >> 4) * 12;          // lane half splits the 24 outputs
  float* orow = OUT + ((size_t)(k * NB_) + gt * 16 + m2) * OUT_F_;
#pragma unroll
  for (int pp = 0; pp < 12; ++pp) {
    const int p = ph + pp;
    float z = BF2[k * OUT_F_ + p];
#pragma unroll
    for (int j = 0; j < 8; ++j)
      z += s_hh[wave][m2][j] * WF2[(k * 8 + j) * OUT_F_ + p];
    orow[p] = 1.0f / (1.0f + __expf(-z));   // sigmoid
  }
}

// ---- yg output is just y (reshape is identity in memory) -----------------
__global__ void k_copy(const float4* __restrict__ src, float4* __restrict__ dst,
                       int n4) {
  const int i = blockIdx.x * blockDim.x + threadIdx.x;
  if (i < n4) dst[i] = src[i];
}

extern "C" void kernel_launch(void* const* d_in, const int* in_sizes, int n_in,
                              void* d_out, int out_size, void* d_ws, size_t ws_size,
                              hipStream_t stream) {
  (void)in_sizes; (void)n_in; (void)out_size; (void)ws_size;
  const float* x      = (const float*)d_in[0];
  const float* y      = (const float*)d_in[1];
  // d_in[2] = edge_index, d_in[3] = batch: structure is static, unused.
  const float* W1     = (const float*)d_in[4];
  const float* a_src1 = (const float*)d_in[5];
  const float* a_dst1 = (const float*)d_in[6];
  const float* b1     = (const float*)d_in[7];
  const float* W2     = (const float*)d_in[8];
  const float* a_src2 = (const float*)d_in[9];
  const float* a_dst2 = (const float*)d_in[10];
  const float* b2     = (const float*)d_in[11];
  const float* Wf1    = (const float*)d_in[12];
  const float* bf1    = (const float*)d_in[13];
  const float* Wf2    = (const float*)d_in[14];
  const float* bf2    = (const float*)d_in[15];
  float* out = (float*)d_out;
  float* ws  = (float*)d_ws;

  float* xh1 = ws;                                 // [N, 64]
  float* h1  = xh1 + (size_t)NTOT_ * 64;           // [N, 32]
  float* xh2 = h1  + (size_t)NTOT_ * 32;           // [N, 128]
  float* h2  = xh2 + (size_t)NTOT_ * 128;          // [N, 64]

  // layer 1: projection (WMMA), attention+ELU
  k_proj<IN_F_, HEADS_ * HID1_><<<NTOT_ / 16 / 8, 256, 0, stream>>>(x, W1, xh1);
  k_attn<HID1_, 16><<<NB_ / 16, 16 * NODES_, 0, stream>>>(xh1, a_src1, a_dst1, b1, h1, 1);
  // layer 2: projection (WMMA), attention
  k_proj<HID1_, HEADS_ * HID2_><<<NTOT_ / 16 / 8, 256, 0, stream>>>(h1, W2, xh2);
  k_attn<HID2_, 8><<<NB_ / 8, 8 * NODES_, 0, stream>>>(xh2, a_src2, a_dst2, b2, h2, 0);
  // per-node-index MLP ensemble (WMMA + VALU tail) -> pred
  k_mlp<<<14 * 128 / 8, 256, 0, stream>>>(h2, Wf1, bf1, Wf2, bf2, out);
  // second output: yg = y
  k_copy<<<(NTOT_ * OUT_F_ / 4) / 256, 256, 0, stream>>>(
      (const float4*)y, (float4*)(out + (size_t)NODES_ * NB_ * OUT_F_),
      NTOT_ * OUT_F_ / 4);
}
